// Block_21895743275611
// MI455X (gfx1250) — matrix operand order repaired and checked by reading, not compiled, not run on hardware
//
#include <hip/hip_runtime.h>
#include <hip/hip_bf16.h>
#include <math.h>

#define T_SEQ 4096
#define C_EMB 1024
#define NH 16
#define HS 64

typedef __attribute__((ext_vector_type(16))) __bf16 v16bf;
typedef __attribute__((ext_vector_type(8))) float v8f;
typedef int v4i_vec __attribute__((vector_size(16)));

union FragBF { unsigned u[8]; v16bf v; };

template <bool B> struct BoolC { static constexpr bool value = B; };

#define ASG __attribute__((address_space(1)))
#define ASL __attribute__((address_space(3)))

#if defined(__AMDGCN__) && __has_builtin(__builtin_amdgcn_global_load_async_to_lds_b128)
#define ASYNC_OK 1
#else
#define ASYNC_OK 0
#endif

template <int N>
__device__ __forceinline__ void wait_async() {
#if ASYNC_OK
#if __has_builtin(__builtin_amdgcn_s_wait_asynccnt)
  __builtin_amdgcn_s_wait_asynccnt((unsigned short)N);
#else
  if (N == 0)
    asm volatile("s_wait_asynccnt 0" ::: "memory");
  else
    asm volatile("s_wait_asynccnt 8" ::: "memory");
#endif
#endif
}

__device__ __forceinline__ unsigned short f2bf(float f) {
  unsigned u = __float_as_uint(f);
  u += 0x7FFFu + ((u >> 16) & 1u);   // round-to-nearest-even
  return (unsigned short)(u >> 16);
}

// ---------------------------------------------------------------------------
// Weight convert + transpose: wt[n][k] = bf16(w[k][n]).  Block (32,8), LDS tile.
// ---------------------------------------------------------------------------
__global__ __launch_bounds__(256) void cvt_transpose(const float* __restrict__ w,
                                                     unsigned short* __restrict__ wt,
                                                     int K, int N) {
  __shared__ float tile[32][33];
  int tx = threadIdx.x, ty = threadIdx.y;
  int n0 = blockIdx.x * 32, k0 = blockIdx.y * 32;
#pragma unroll
  for (int i = 0; i < 4; ++i)
    tile[ty + i * 8][tx] = w[(size_t)(k0 + ty + i * 8) * N + (n0 + tx)];
  __syncthreads();
#pragma unroll
  for (int i = 0; i < 4; ++i)
    wt[(size_t)(n0 + ty + i * 8) * K + (k0 + tx)] = f2bf(tile[tx][ty + i * 8]);
}

// ---------------------------------------------------------------------------
// LayerNorm row kernel: one block per row of 1024, output bf16.
// ---------------------------------------------------------------------------
__global__ __launch_bounds__(256) void ln_kernel(const float* __restrict__ x,
                                                 const float* __restrict__ g,
                                                 unsigned short* __restrict__ out) {
  int row = blockIdx.x;
  const float* xr = x + (size_t)row * C_EMB;
  float v[4];
  float s = 0.f, s2 = 0.f;
#pragma unroll
  for (int i = 0; i < 4; ++i) {
    v[i] = xr[threadIdx.x + i * 256];
    s += v[i];
    s2 += v[i] * v[i];
  }
  __shared__ float r1[256], r2[256];
  r1[threadIdx.x] = s;
  r2[threadIdx.x] = s2;
  __syncthreads();
  for (int off = 128; off > 0; off >>= 1) {
    if (threadIdx.x < off) {
      r1[threadIdx.x] += r1[threadIdx.x + off];
      r2[threadIdx.x] += r2[threadIdx.x + off];
    }
    __syncthreads();
  }
  float mu = r1[0] * (1.f / C_EMB);
  float var = r2[0] * (1.f / C_EMB) - mu * mu;
  float rstd = rsqrtf(var + 1e-5f);
#pragma unroll
  for (int i = 0; i < 4; ++i) {
    int c = threadIdx.x + i * 256;
    out[(size_t)row * C_EMB + c] = f2bf((v[i] - mu) * rstd * g[c]);
  }
}

// ---------------------------------------------------------------------------
// LDS fragment loaders for v_wmma_f32_16x16x32_bf16 (stride 72 elements).
// A (16x32): lane m=l&15, half h=l>>4; VGPR v holds K = (v>>2)*16 + h*8 + (v&3)*2, +1
// B (32x16): lane n=l&15, half h;      VGPR v holds K = (v>>2)*16 + h*8 + (v&3)*2, +1
// ---------------------------------------------------------------------------
__device__ __forceinline__ v16bf frag_a_lds(const unsigned short* p, int lane) {
  FragBF f;
  int m = lane & 15, hh = lane >> 4;
  const unsigned short* q = p + m * 72;
#pragma unroll
  for (int v = 0; v < 8; ++v) {
    int k = ((v >> 2) << 4) + (hh << 3) + ((v & 3) << 1);
    f.u[v] = *(const unsigned*)(q + k);
  }
  return f.v;
}
__device__ __forceinline__ v16bf frag_b_lds(const unsigned short* p, int lane) {
  FragBF f;
  int n = lane & 15, hh = lane >> 4;
  const unsigned short* q = p + n * 72;
#pragma unroll
  for (int v = 0; v < 8; ++v) {
    int k = ((v >> 2) << 4) + (hh << 3) + ((v & 3) << 1);
    f.u[v] = *(const unsigned*)(q + k);
  }
  return f.v;
}

// ---------------------------------------------------------------------------
// Generic bf16 GEMM:  out[M,N] = A[M,K] * Bt[N,K]^T  (Bt stored N-major)
// Block tile 128x128, BK=64, 8 waves (4x2), per-wave 32x64 -> 8 v8f accums.
// Double-buffered LDS fed by GLOBAL_LOAD_ASYNC_TO_LDS_B128 (ASYNCcnt-tracked),
// so the next K-slab streams into LDS while WMMAs consume the current one.
// EPI 0: QKV scatter (q scaled 1/8, v transposed)   EPI 1: GELU->bf16
// EPI 2: +residual -> f32
// ---------------------------------------------------------------------------
template <int EPI>
__global__ __launch_bounds__(256) void gemm_bf16(
    const unsigned short* __restrict__ A, const unsigned short* __restrict__ Bt,
    int M, int N, int K, float* __restrict__ outf,
    unsigned short* __restrict__ outb, const float* __restrict__ res,
    unsigned short* __restrict__ qb, unsigned short* __restrict__ kb2,
    unsigned short* __restrict__ vb) {
  __shared__ unsigned short lsa[2][128 * 72];
  __shared__ unsigned short lsb[2][128 * 72];
  int tid = threadIdx.x;
  int lane = tid & 31, wid = tid >> 5;
  int m0 = blockIdx.y * 128, n0 = blockIdx.x * 128;
  int wm = (wid >> 1) * 32, wn = (wid & 1) * 64;

  v8f acc[2][4] = {};

  int rr[4], cc[4];
#pragma unroll
  for (int i = 0; i < 4; ++i) {
    int e = (tid + i * 256) * 8;
    rr[i] = e >> 6;
    cc[i] = e & 63;
  }

  auto stage = [&](int kk, int buf) {
#pragma unroll
    for (int i = 0; i < 4; ++i) {
      const unsigned short* ga = A + (size_t)(m0 + rr[i]) * K + kk + cc[i];
      const unsigned short* gb = Bt + (size_t)(n0 + rr[i]) * K + kk + cc[i];
      unsigned short* la = &lsa[buf][rr[i] * 72 + cc[i]];
      unsigned short* lb = &lsb[buf][rr[i] * 72 + cc[i]];
#if ASYNC_OK
      __builtin_amdgcn_global_load_async_to_lds_b128(
          (ASG v4i_vec*)(ASG void*)ga, (ASL v4i_vec*)(ASL void*)la, 0, 0);
      __builtin_amdgcn_global_load_async_to_lds_b128(
          (ASG v4i_vec*)(ASG void*)gb, (ASL v4i_vec*)(ASL void*)lb, 0, 0);
#else
      *(uint4*)la = *(const uint4*)ga;
      *(uint4*)lb = *(const uint4*)gb;
#endif
    }
  };

  int nstage = K >> 6;
  stage(0, 0);
  for (int s = 0; s < nstage; ++s) {
    int cur = s & 1;
    if (s + 1 < nstage) {
      stage((s + 1) * 64, cur ^ 1);  // overlap next slab with this slab's WMMAs
      wait_async<8>();               // only current slab's 8 loads must be done
    } else {
      wait_async<0>();
    }
    __syncthreads();
#pragma unroll
    for (int t = 0; t < 2; ++t) {
      // hoist all fragment loads into distinct registers so ds_loads overlap WMMAs
      v16bf a0 = frag_a_lds(&lsa[cur][(wm + 0) * 72 + t * 32], lane);
      v16bf a1 = frag_a_lds(&lsa[cur][(wm + 16) * 72 + t * 32], lane);
      v16bf bfr[4];
#pragma unroll
      for (int j = 0; j < 4; ++j)
        bfr[j] = frag_b_lds(&lsb[cur][(wn + j * 16) * 72 + t * 32], lane);
#pragma unroll
      for (int j = 0; j < 4; ++j) {
        acc[0][j] = __builtin_amdgcn_wmma_f32_16x16x32_bf16(
            false, a0, false, bfr[j], (short)0, acc[0][j], false, false);
        acc[1][j] = __builtin_amdgcn_wmma_f32_16x16x32_bf16(
            false, a1, false, bfr[j], (short)0, acc[1][j], false, false);
      }
    }
    __syncthreads();
  }

  int hh = lane >> 4, nl = lane & 15;
#pragma unroll
  for (int i = 0; i < 2; ++i)
#pragma unroll
    for (int j = 0; j < 4; ++j)
#pragma unroll
      for (int v = 0; v < 8; ++v) {
        int row = m0 + wm + i * 16 + v + 8 * hh;
        int col = n0 + wn + j * 16 + nl;
        float val = acc[i][j][v];
        if (EPI == 0) {
          if (col < C_EMB) {
            int head = col >> 6, hs = col & 63;
            qb[((size_t)head * T_SEQ + row) * HS + hs] = f2bf(val * 0.125f);
          } else if (col < 2 * C_EMB) {
            int c2 = col - C_EMB;
            int head = c2 >> 6, hs = c2 & 63;
            kb2[((size_t)head * T_SEQ + row) * HS + hs] = f2bf(val);
          } else {
            int c2 = col - 2 * C_EMB;
            int head = c2 >> 6, hs = c2 & 63;
            vb[((size_t)head * HS + hs) * T_SEQ + row] = f2bf(val);
          }
        } else if (EPI == 1) {
          float gl = 0.5f * val * (1.0f + erff(val * 0.70710678118654752f));
          outb[(size_t)row * N + col] = f2bf(gl);
        } else {
          outf[(size_t)row * N + col] = val + res[(size_t)row * N + col];
        }
      }
}

// ---------------------------------------------------------------------------
// Flash attention: grid (T/128, NH), 8 waves/block, each wave owns 16 q rows.
// q,k: [H][T][hs] bf16 (q pre-scaled 1/sqrt(hs)); v: [H][hs][T] bf16.
// Online softmax in fp32; P converted C-layout -> A-layout via per-wave LDS.
// Causal loop is split into a no-mask phase (bulk) and a masked tail phase.
// ---------------------------------------------------------------------------
__global__ __launch_bounds__(256) void attn_kernel(
    const unsigned short* __restrict__ qb, const unsigned short* __restrict__ kb,
    const unsigned short* __restrict__ vb, unsigned short* __restrict__ attn_out) {
  __shared__ unsigned short pshm[8][16 * 40];
  int tid = threadIdx.x, lane = tid & 31, wid = tid >> 5;
  int head = blockIdx.y;
  int qbase = blockIdx.x * 128 + wid * 16;
  int nl = lane & 15, hh = lane >> 4;

  const unsigned short* qh = qb + (size_t)head * T_SEQ * HS;
  const unsigned short* kh = kb + (size_t)head * T_SEQ * HS;
  const unsigned short* vh = vb + (size_t)head * HS * T_SEQ;

  FragBF qf[2];
#pragma unroll
  for (int t = 0; t < 2; ++t)
#pragma unroll
    for (int v = 0; v < 8; ++v) {
      int k = t * 32 + ((v >> 2) << 4) + (hh << 3) + ((v & 3) << 1);
      qf[t].u[v] = *(const unsigned*)(qh + (size_t)(qbase + nl) * HS + k);
    }

  v8f o[4] = {};
  float ms[8], ls[8];
#pragma unroll
  for (int v = 0; v < 8; ++v) { ms[v] = -1e30f; ls[v] = 0.f; }

  int kbmax = (qbase + 15) >> 5;   // last key block index (inclusive)
  int nfull = (qbase + 1) >> 5;    // blocks [0,nfull) need no causal mask

  auto process = [&](int k0, auto maskC) {
    constexpr bool MASKED = decltype(maskC)::value;
    if (k0 + 32 <= kbmax * 32) {  // prefetch next key/value block
      __builtin_prefetch(kh + (size_t)(k0 + 32 + nl) * HS, 0, 3);
      __builtin_prefetch(vh + (size_t)(nl * 4) * T_SEQ + k0 + 32, 0, 3);
    }
    // ---- load all K fragments up front (independent regs -> overlapped loads)
    FragBF kf[2][2];
#pragma unroll
    for (int j = 0; j < 2; ++j) {
      int key = k0 + j * 16 + nl;
#pragma unroll
      for (int t = 0; t < 2; ++t)
#pragma unroll
        for (int v = 0; v < 8; ++v) {
          int kkk = t * 32 + ((v >> 2) << 4) + (hh << 3) + ((v & 3) << 1);
          kf[j][t].u[v] = *(const unsigned*)(kh + (size_t)key * HS + kkk);
        }
    }
    v8f s[2] = {};
#pragma unroll
    for (int j = 0; j < 2; ++j)
#pragma unroll
      for (int t = 0; t < 2; ++t)
        s[j] = __builtin_amdgcn_wmma_f32_16x16x32_bf16(
            false, qf[t].v, false, kf[j][t].v, (short)0, s[j], false, false);

    // ---- load all V fragments up front too (they don't depend on P)
    FragBF vf[4];
#pragma unroll
    for (int jn = 0; jn < 4; ++jn) {
      int hscol = jn * 16 + nl;
#pragma unroll
      for (int v = 0; v < 8; ++v) {
        int key = k0 + ((v >> 2) << 4) + (hh << 3) + ((v & 3) << 1);
        vf[jn].u[v] = *(const unsigned*)(vh + (size_t)hscol * T_SEQ + key);
      }
    }

    // ---- online softmax per row (C layout: lane holds col nl, rows v+8*hh)
#pragma unroll
    for (int v = 0; v < 8; ++v) {
      float s0 = s[0][v], s1 = s[1][v];
      if (MASKED) {
        int row = qbase + v + 8 * hh;
        s0 = (k0 + nl <= row) ? s0 : -1e30f;
        s1 = (k0 + 16 + nl <= row) ? s1 : -1e30f;
      }
      float mx = fmaxf(s0, s1);
#pragma unroll
      for (int off = 1; off < 16; off <<= 1)
        mx = fmaxf(mx, __shfl_xor(mx, off, 32));
      float mnew = fmaxf(ms[v], mx);
      float alpha = __expf(ms[v] - mnew);
      float p0 = __expf(s0 - mnew);
      float p1 = __expf(s1 - mnew);
      float rs = p0 + p1;
#pragma unroll
      for (int off = 1; off < 16; off <<= 1)
        rs += __shfl_xor(rs, off, 32);
      ls[v] = ls[v] * alpha + rs;
      ms[v] = mnew;
#pragma unroll
      for (int jn = 0; jn < 4; ++jn) o[jn][v] *= alpha;
      pshm[wid][(v + 8 * hh) * 40 + nl] = f2bf(p0);
      pshm[wid][(v + 8 * hh) * 40 + 16 + nl] = f2bf(p1);
    }
    asm volatile("s_wait_dscnt 0" ::: "memory");  // per-wave LDS RAW fence
    FragBF pf;
#pragma unroll
    for (int v = 0; v < 8; ++v) {
      int kkk = ((v >> 2) << 4) + (hh << 3) + ((v & 3) << 1);
      pf.u[v] = *(const unsigned*)(&pshm[wid][nl * 40 + kkk]);
    }
#pragma unroll
    for (int jn = 0; jn < 4; ++jn)
      o[jn] = __builtin_amdgcn_wmma_f32_16x16x32_bf16(
          false, pf.v, false, vf[jn].v, (short)0, o[jn], false, false);
  };

  for (int kbi = 0; kbi < nfull; ++kbi) process(kbi * 32, BoolC<false>{});
  for (int kbi = nfull; kbi <= kbmax; ++kbi) process(kbi * 32, BoolC<true>{});

#pragma unroll
  for (int v = 0; v < 8; ++v) {
    float inv = 1.f / ls[v];
    int row = qbase + v + 8 * hh;
#pragma unroll
    for (int jn = 0; jn < 4; ++jn) {
      int col = head * HS + jn * 16 + nl;
      attn_out[(size_t)row * C_EMB + col] = f2bf(o[jn][v] * inv);
    }
  }
}

// ---------------------------------------------------------------------------
extern "C" void kernel_launch(void* const* d_in, const int* in_sizes, int n_in,
                              void* d_out, int out_size, void* d_ws, size_t ws_size,
                              hipStream_t stream) {
  (void)in_sizes; (void)n_in; (void)out_size; (void)ws_size;
  const float* x    = (const float*)d_in[0];
  const float* wqkv = (const float*)d_in[1];
  const float* wap  = (const float*)d_in[2];
  const float* wfc  = (const float*)d_in[3];
  const float* wmp  = (const float*)d_in[4];
  const float* ln1w = (const float*)d_in[5];
  const float* ln2w = (const float*)d_in[6];
  float* out = (float*)d_out;

  char* p = (char*)d_ws;
  auto alloc = [&](size_t bytes) {
    void* r = (void*)p;
    p += (bytes + 255) & ~(size_t)255;
    return r;
  };
  unsigned short* wqkv_t = (unsigned short*)alloc((size_t)3 * C_EMB * C_EMB * 2);
  unsigned short* wap_t  = (unsigned short*)alloc((size_t)C_EMB * C_EMB * 2);
  unsigned short* wfc_t  = (unsigned short*)alloc((size_t)4 * C_EMB * C_EMB * 2);
  unsigned short* wmp_t  = (unsigned short*)alloc((size_t)4 * C_EMB * C_EMB * 2);
  unsigned short* hbuf   = (unsigned short*)alloc((size_t)T_SEQ * C_EMB * 2);
  unsigned short* qbuf   = (unsigned short*)alloc((size_t)T_SEQ * C_EMB * 2);
  unsigned short* kbuf   = (unsigned short*)alloc((size_t)T_SEQ * C_EMB * 2);
  unsigned short* vbuf   = (unsigned short*)alloc((size_t)T_SEQ * C_EMB * 2);
  unsigned short* aout   = (unsigned short*)alloc((size_t)T_SEQ * C_EMB * 2);
  float*          x2     = (float*)alloc((size_t)T_SEQ * C_EMB * 4);
  unsigned short* h2buf  = (unsigned short*)alloc((size_t)T_SEQ * C_EMB * 2);
  unsigned short* act    = (unsigned short*)alloc((size_t)T_SEQ * 4 * C_EMB * 2);

  dim3 tb(32, 8);
  cvt_transpose<<<dim3(3 * C_EMB / 32, C_EMB / 32), tb, 0, stream>>>(wqkv, wqkv_t, C_EMB, 3 * C_EMB);
  cvt_transpose<<<dim3(C_EMB / 32, C_EMB / 32), tb, 0, stream>>>(wap, wap_t, C_EMB, C_EMB);
  cvt_transpose<<<dim3(4 * C_EMB / 32, C_EMB / 32), tb, 0, stream>>>(wfc, wfc_t, C_EMB, 4 * C_EMB);
  cvt_transpose<<<dim3(C_EMB / 32, 4 * C_EMB / 32), tb, 0, stream>>>(wmp, wmp_t, 4 * C_EMB, C_EMB);

  ln_kernel<<<T_SEQ, 256, 0, stream>>>(x, ln1w, hbuf);

  gemm_bf16<0><<<dim3(3 * C_EMB / 128, T_SEQ / 128), 256, 0, stream>>>(
      hbuf, wqkv_t, T_SEQ, 3 * C_EMB, C_EMB, nullptr, nullptr, nullptr, qbuf, kbuf, vbuf);

  attn_kernel<<<dim3(T_SEQ / 128, NH), 256, 0, stream>>>(qbuf, kbuf, vbuf, aout);

  gemm_bf16<2><<<dim3(C_EMB / 128, T_SEQ / 128), 256, 0, stream>>>(
      aout, wap_t, T_SEQ, C_EMB, C_EMB, x2, nullptr, x, nullptr, nullptr, nullptr);

  ln_kernel<<<T_SEQ, 256, 0, stream>>>(x2, ln2w, h2buf);

  gemm_bf16<1><<<dim3(4 * C_EMB / 128, T_SEQ / 128), 256, 0, stream>>>(
      h2buf, wfc_t, T_SEQ, 4 * C_EMB, C_EMB, nullptr, act, nullptr, nullptr, nullptr, nullptr);

  gemm_bf16<2><<<dim3(C_EMB / 128, T_SEQ / 128), 256, 0, stream>>>(
      act, wmp_t, T_SEQ, C_EMB, 4 * C_EMB, out, nullptr, x2, nullptr, nullptr, nullptr);
}
